// GatedNNMF_54829552500941
// MI455X (gfx1250) — matrix operand order repaired and checked by reading, not compiled, not run on hardware
//
#include <hip/hip_runtime.h>
#include <hip/hip_bf16.h>
#include <math.h>

// ---------------------------------------------------------------------------
// Gated NMF FFN for MI455X (gfx1250), wave32 + v_wmma_f32_16x16x32_bf16.
// Big GEMMs stage the block-shared B panel into LDS with the Tensor Data
// Mover (tensor_load_to_lds + s_wait_tensorcnt); small NMF GEMMs read
// straight through L2. All GEMM operands are bf16 with K contiguous (NT).
// ---------------------------------------------------------------------------

typedef __bf16 bf16_t;
typedef __attribute__((ext_vector_type(16))) __bf16 v16bf;
typedef __attribute__((ext_vector_type(8)))  __bf16 v8bf;
typedef __attribute__((ext_vector_type(8)))  float  v8f;

typedef __attribute__((ext_vector_type(4))) unsigned int tdm_u4;
typedef __attribute__((ext_vector_type(8))) int          tdm_i8;
typedef __attribute__((ext_vector_type(4))) int          tdm_i4;

#define NMF_EPS 1e-6f
#define LN_EPS  1e-5f

#if defined(__has_builtin)
#if __has_builtin(__builtin_amdgcn_tensor_load_to_lds)
#define HAVE_TDM 1
#endif
#endif

__device__ __forceinline__ float wredsum(float v) {
#pragma unroll
  for (int o = 16; o > 0; o >>= 1) v += __shfl_xor(v, o, 32);
  return v;
}
__device__ __forceinline__ float wredmax(float v) {
#pragma unroll
  for (int o = 16; o > 0; o >>= 1) v = fmaxf(v, __shfl_xor(v, o, 32));
  return v;
}

#ifdef HAVE_TDM
// TDM 2D tile load: rows x cols (bf16) from global (row stride ldb elems)
// into LDS at byte offset lds_off, rows packed at cols*2 bytes each.
// D# layout per CDNA5 ISA ch.8 (group0: count/lds/global/type, group1: dims).
// This toolchain declares the 6-arg builtin:
//   (u32x4 g0, i32x8 g1, i32x4 g2, i32x4 g3, i32x8 pad, i32 cpol)
__device__ __forceinline__ void tdm_load_b_tile(unsigned lds_off, const bf16_t* gptr,
                                                long ldb, int rows, int cols) {
  const unsigned long long ga = (unsigned long long)(size_t)gptr;
  tdm_u4 g0;
  g0[0] = 1u;                                              // count=1 (valid), user desc
  g0[1] = lds_off;                                         // lds_addr (bytes)
  g0[2] = (unsigned)(ga & 0xFFFFFFFFu);                    // global_addr[31:0]
  g0[3] = (unsigned)((ga >> 32) & 0x01FFFFFFu) | (2u << 30); // addr[56:32] | type=2
  const unsigned uc = (unsigned)cols, ur = (unsigned)rows;
  const unsigned long long st = (unsigned long long)ldb;   // dim0 stride (elems)
  tdm_i8 g1;
  g1[0] = (int)(1u << 16);                                 // data_size=1 -> 2 bytes
  g1[1] = (int)((uc & 0xFFFFu) << 16);                     // tensor_dim0[15:0]
  g1[2] = (int)((uc >> 16) | ((ur & 0xFFFFu) << 16));      // td0[31:16] | td1[15:0]
  g1[3] = (int)((ur >> 16) | ((uc & 0xFFFFu) << 16));      // td1[31:16] | tile_dim0
  g1[4] = (int)(ur & 0xFFFFu);                             // tile_dim1 (tile_dim2=0)
  g1[5] = (int)(unsigned)(st & 0xFFFFFFFFu);               // dim0_stride[31:0]
  g1[6] = (int)(unsigned)((st >> 32) & 0xFFFFu);           // dim0_stride[47:32]
  g1[7] = 0;
  const tdm_i4 gz4 = {0, 0, 0, 0};
  const tdm_i8 gz8 = {0, 0, 0, 0, 0, 0, 0, 0};
  __builtin_amdgcn_tensor_load_to_lds(g0, g1, gz4, gz4, gz8, 0);
}
#endif

// Cooperative fallback loader (256 threads), 16B per access.
__device__ __forceinline__ void coop_load_b_tile(bf16_t* shB, const bf16_t* gptr,
                                                 long ldb, int rows, int cols) {
  const int vecPerRow = cols >> 3;            // 8 bf16 per 16B
  const int total = rows * vecPerRow;
  for (int i = threadIdx.x; i < total; i += 256) {
    const int r = i / vecPerRow;
    const int c = (i - r * vecPerRow) << 3;
    *(uint4*)(shB + (size_t)r * cols + c) = *(const uint4*)(gptr + (size_t)r * ldb + c);
  }
}

// ---------------------------------------------------------------------------
// Direct-from-global WMMA mainloop: one wave makes a 16x64 fp32 tile of
// C = A * B^T.  A:[M,K] bf16 K-contig, B:[N,K] bf16 K-contig.
// Fragments use contiguous k = 16*(lane/16) + s for A and B (16-bit A 16x32, B 32x16, f32 C/D).
// ---------------------------------------------------------------------------
__device__ __forceinline__ void wmma_loop(const bf16_t* __restrict__ A, long lda,
                                          const bf16_t* __restrict__ B, long ldb,
                                          int K, int mBase, int nBase, v8f acc[4]) {
  const int lane = threadIdx.x & 31;
  const int lh   = lane & 15;
  const int sel  = lane >> 4;
  const bf16_t* aptr = A + (size_t)(mBase + lh) * lda + 16 * sel;
  const bf16_t* bptr = B + (size_t)(nBase + lh) * ldb + 16 * sel;
  for (int k = 0; k < K; k += 32) {
    v8bf alo = *(const v8bf*)(aptr + k);
    v8bf ahi = *(const v8bf*)(aptr + k + 8);
    v16bf a;
#pragma unroll
    for (int i = 0; i < 8; ++i) { a[i] = alo[i]; a[i + 8] = ahi[i]; }
    __builtin_prefetch(aptr + k + 256, 0, 1);   // global_prefetch_b8
#pragma unroll
    for (int j = 0; j < 4; ++j) {
      v16bf b = *(const v16bf*)(bptr + (size_t)(16 * j) * ldb + k);
      acc[j] = __builtin_amdgcn_wmma_f32_16x16x32_bf16(
          false, a, false, b, (short)0, acc[j], false, false);
    }
  }
}

// ---------------------------------------------------------------------------
// LDS-staged WMMA mainloop for the big GEMMs: per-block 64xKC B panel staged
// through the TDM (wave 0 issues, s_wait_tensorcnt, barrier), waves read B
// fragments via ds_load. Cuts B L2 traffic 8x (8 waves share the panel).
// Must be called by all 8 waves of the block (contains barriers).
// ---------------------------------------------------------------------------
template <int KC>
__device__ __forceinline__ void wmma_loop_ldsB(const bf16_t* __restrict__ A, long lda,
                                               const bf16_t* __restrict__ B, long ldb,
                                               int K, int mBase, int nBase,
                                               bf16_t* shB, v8f acc[4]) {
  const int lane = threadIdx.x & 31;
  const int lh   = lane & 15;
  const int sel  = lane >> 4;
  const bf16_t* aptr = A + (size_t)(mBase + lh) * lda + 16 * sel;
  const bf16_t* brow = B + (size_t)nBase * ldb;
  for (int kc = 0; kc < K; kc += KC) {
    __syncthreads();                       // panel consumed by all waves
#ifdef HAVE_TDM
    if (threadIdx.x < 32) {                // TDM issue is per-wave, EXEC-ignored
      tdm_load_b_tile((unsigned)(size_t)shB, brow + kc, ldb, 64, KC);
      __builtin_amdgcn_s_wait_tensorcnt(0);
    }
#else
    coop_load_b_tile(shB, brow + kc, ldb, 64, KC);
#endif
    __syncthreads();                       // panel visible to all waves
#pragma unroll 2
    for (int k2 = 0; k2 < KC; k2 += 32) {
      v8bf alo = *(const v8bf*)(aptr + kc + k2);
      v8bf ahi = *(const v8bf*)(aptr + kc + k2 + 8);
      v16bf a;
#pragma unroll
      for (int i = 0; i < 8; ++i) { a[i] = alo[i]; a[i + 8] = ahi[i]; }
      __builtin_prefetch(aptr + kc + k2 + 512, 0, 1);
#pragma unroll
      for (int j = 0; j < 4; ++j) {
        const bf16_t* bl = shB + (size_t)(16 * j + lh) * KC + k2 + 16 * sel;
        v16bf b = *(const v16bf*)bl;       // ds_load_b128 x2
        acc[j] = __builtin_amdgcn_wmma_f32_16x16x32_bf16(
            false, a, false, b, (short)0, acc[j], false, false);
      }
    }
  }
}

// ---------------- generic batched GEMMs (plain fp32 / bf16 output) ----------
__global__ void k_gemm_f32(const bf16_t* A, long lda, long sA,
                           const bf16_t* B, long ldb, long sB,
                           float* C, long ldc, long sC, int M, int K) {
  const int wave  = threadIdx.x >> 5;
  const int mBase = (blockIdx.y * 8 + wave) * 16;
  if (mBase >= M) return;                       // wave-uniform early out
  const int nBase = blockIdx.x * 64;
  const int bz = blockIdx.z;
  A += (size_t)bz * sA; B += (size_t)bz * sB; C += (size_t)bz * sC;
  v8f acc[4] = {};
  wmma_loop(A, lda, B, ldb, K, mBase, nBase, acc);
  const int lane = threadIdx.x & 31, lh = lane & 15, sel = lane >> 4;
#pragma unroll
  for (int j = 0; j < 4; ++j)
#pragma unroll
    for (int v = 0; v < 8; ++v)
      C[(size_t)(mBase + v + 8 * sel) * ldc + nBase + 16 * j + lh] = acc[j][v];
}

__global__ void k_gemm_bf16(const bf16_t* A, long lda, long sA,
                            const bf16_t* B, long ldb, long sB,
                            bf16_t* C, long ldc, long sC, int M, int K) {
  const int wave  = threadIdx.x >> 5;
  const int mBase = (blockIdx.y * 8 + wave) * 16;
  if (mBase >= M) return;
  const int nBase = blockIdx.x * 64;
  const int bz = blockIdx.z;
  A += (size_t)bz * sA; B += (size_t)bz * sB; C += (size_t)bz * sC;
  v8f acc[4] = {};
  wmma_loop(A, lda, B, ldb, K, mBase, nBase, acc);
  const int lane = threadIdx.x & 31, lh = lane & 15, sel = lane >> 4;
#pragma unroll
  for (int j = 0; j < 4; ++j)
#pragma unroll
    for (int v = 0; v < 8; ++v)
      C[(size_t)(mBase + v + 8 * sel) * ldc + nBase + 16 * j + lh] = (bf16_t)acc[j][v];
}

// ---------------- GEMM1: h = gelu(LN(x) @ Uw^T + Ub), split z1/z2 -----------
__global__ void k_gemm1(const bf16_t* __restrict__ A, const bf16_t* __restrict__ B,
                        const float* __restrict__ bias,
                        bf16_t* __restrict__ z1, float* __restrict__ z2) {
  __shared__ bf16_t shB[64 * 256];
  const int wave  = threadIdx.x >> 5;
  const int mBase = (blockIdx.y * 8 + wave) * 16;
  const int nBase = blockIdx.x * 64;
  v8f acc[4] = {};
  wmma_loop_ldsB<256>(A, 1024, B, 1024, 1024, mBase, nBase, shB, acc);
  const int lane = threadIdx.x & 31, lh = lane & 15, sel = lane >> 4;
#pragma unroll
  for (int j = 0; j < 4; ++j) {
    const int col = nBase + 16 * j + lh;
    const float bv = bias[col];
#pragma unroll
    for (int v = 0; v < 8; ++v) {
      const int row = mBase + v + 8 * sel;
      float val = acc[j][v] + bv;
      val = 0.5f * val * (1.0f + erff(val * 0.70710678118654752f)); // exact GELU
      if (col < 2048) z1[(size_t)row * 2048 + col] = (bf16_t)val;
      else            z2[(size_t)row * 2048 + (col - 2048)] = val;
    }
  }
}

// ---------------- reconstruction: g = z1 * (bases @ coef^T) -----------------
__global__ void k_recon(const bf16_t* __restrict__ basesb, const bf16_t* __restrict__ coefb,
                        const bf16_t* __restrict__ z1, bf16_t* __restrict__ g) {
  const int wave  = threadIdx.x >> 5;
  const int mBase = (blockIdx.y * 8 + wave) * 16;   // d
  const int nBase = blockIdx.x * 64;                // n
  const int bz = blockIdx.z;
  const bf16_t* A = basesb + (size_t)bz * 1024 * 64;
  const bf16_t* B = coefb  + (size_t)bz * 2048 * 64;
  v8f acc[4] = {};
  wmma_loop(A, 64, B, 64, 64, mBase, nBase, acc);
  const int lane = threadIdx.x & 31, lh = lane & 15, sel = lane >> 4;
#pragma unroll
  for (int j = 0; j < 4; ++j) {
    const int n = nBase + 16 * j + lh;
#pragma unroll
    for (int v = 0; v < 8; ++v) {
      const int d = mBase + v + 8 * sel;
      const size_t gi = ((size_t)bz * 1024 + d) * 2048 + n;
      g[gi] = (bf16_t)((float)z1[gi] * acc[j][v]);
    }
  }
}

// ---------------- final: out = g @ Vw^T + Vb --------------------------------
__global__ void k_final(const bf16_t* __restrict__ g, const bf16_t* __restrict__ Vwb,
                        const float* __restrict__ bias, float* __restrict__ out) {
  __shared__ bf16_t shB[64 * 256];
  const int wave  = threadIdx.x >> 5;
  const int mBase = (blockIdx.y * 8 + wave) * 16;
  const int nBase = blockIdx.x * 64;
  v8f acc[4] = {};
  wmma_loop_ldsB<256>(g, 2048, Vwb, 2048, 2048, mBase, nBase, shB, acc);
  const int lane = threadIdx.x & 31, lh = lane & 15, sel = lane >> 4;
#pragma unroll
  for (int j = 0; j < 4; ++j) {
    const int col = nBase + 16 * j + lh;
    const float bv = bias[col];
#pragma unroll
    for (int v = 0; v < 8; ++v)
      out[(size_t)(mBase + v + 8 * sel) * 1024 + col] = acc[j][v] + bv;
  }
}

// ---------------- LayerNorm 1: x[16384,1024] -> bf16 ------------------------
__global__ void k_ln1(const float* __restrict__ x, const float* __restrict__ gam,
                      const float* __restrict__ bet, bf16_t* __restrict__ outp) {
  const int row = blockIdx.x;
  const float* xr = x + (size_t)row * 1024;
  float v[4]; float s = 0.f, ss = 0.f;
#pragma unroll
  for (int i = 0; i < 4; ++i) { v[i] = xr[threadIdx.x + 256 * i]; s += v[i]; ss += v[i] * v[i]; }
  s = wredsum(s); ss = wredsum(ss);
  __shared__ float sh[16];
  const int wave = threadIdx.x >> 5, lane = threadIdx.x & 31;
  if (lane == 0) { sh[wave] = s; sh[8 + wave] = ss; }
  __syncthreads();
  if (threadIdx.x == 0) {
    float ts = 0.f, tss = 0.f;
    for (int w = 0; w < 8; ++w) { ts += sh[w]; tss += sh[8 + w]; }
    sh[0] = ts * (1.f / 1024.f);
    sh[1] = tss * (1.f / 1024.f);
  }
  __syncthreads();
  const float mean = sh[0];
  const float inv  = rsqrtf(sh[1] - mean * mean + LN_EPS);
#pragma unroll
  for (int i = 0; i < 4; ++i) {
    const int c = threadIdx.x + 256 * i;
    outp[(size_t)row * 1024 + c] = (bf16_t)((v[i] - mean) * inv * gam[c] + bet[c]);
  }
}

// ---------------- LayerNorm 2 + ReLU -> xn [m,2048] and xnT [b,2048,1024] ---
__global__ void k_ln2(const float* __restrict__ z2, const float* __restrict__ gam,
                      const float* __restrict__ bet,
                      bf16_t* __restrict__ xn, bf16_t* __restrict__ xnT) {
  const int row = blockIdx.x;                  // m = b*1024 + d
  const float* zr = z2 + (size_t)row * 2048;
  float v[8]; float s = 0.f, ss = 0.f;
#pragma unroll
  for (int i = 0; i < 8; ++i) { v[i] = zr[threadIdx.x + 256 * i]; s += v[i]; ss += v[i] * v[i]; }
  s = wredsum(s); ss = wredsum(ss);
  __shared__ float sh[16];
  const int wave = threadIdx.x >> 5, lane = threadIdx.x & 31;
  if (lane == 0) { sh[wave] = s; sh[8 + wave] = ss; }
  __syncthreads();
  if (threadIdx.x == 0) {
    float ts = 0.f, tss = 0.f;
    for (int w = 0; w < 8; ++w) { ts += sh[w]; tss += sh[8 + w]; }
    sh[0] = ts * (1.f / 2048.f);
    sh[1] = tss * (1.f / 2048.f);
  }
  __syncthreads();
  const float mean = sh[0];
  const float inv  = rsqrtf(sh[1] - mean * mean + LN_EPS);
  const int b = row >> 10, d = row & 1023;
#pragma unroll
  for (int i = 0; i < 8; ++i) {
    const int c = threadIdx.x + 256 * i;
    float y = fmaxf((v[i] - mean) * inv * gam[c] + bet[c], 0.0f);
    const bf16_t yb = (bf16_t)y;
    xn[(size_t)row * 2048 + c] = yb;
    xnT[((size_t)b * 2048 + c) * 1024 + d] = yb;
  }
}

// ---------------- coef0 softmax over R=64 (one wave per row) ----------------
__global__ void k_softmax(const float* __restrict__ num, float* __restrict__ coef,
                          bf16_t* __restrict__ cb, bf16_t* __restrict__ cT) {
  const int row  = blockIdx.x * 8 + (threadIdx.x >> 5);   // [0, 16*2048)
  const int lane = threadIdx.x & 31;
  const float* nr = num + (size_t)row * 64;
  const float a = nr[lane], c2 = nr[lane + 32];
  const float mx = wredmax(fmaxf(a, c2));
  const float ea = __expf(a - mx), eb = __expf(c2 - mx);
  const float inv = 1.0f / wredsum(ea + eb);
  const float pa = ea * inv, pb = eb * inv;
  const int b = row >> 11, n = row & 2047;
  const size_t base = (size_t)row * 64;
  coef[base + lane] = pa;       coef[base + lane + 32] = pb;
  cb[base + lane]   = (bf16_t)pa; cb[base + lane + 32] = (bf16_t)pb;
  const size_t tb = (size_t)b * 64 * 2048;
  cT[tb + (size_t)lane * 2048 + n]        = (bf16_t)pa;
  cT[tb + (size_t)(lane + 32) * 2048 + n] = (bf16_t)pb;
}

// ---------------- multiplicative updates ------------------------------------
__global__ void k_update_coef(float* __restrict__ coef, const float* __restrict__ num,
                              const float* __restrict__ den,
                              bf16_t* __restrict__ cb, bf16_t* __restrict__ cT, int n) {
  const int i = blockIdx.x * 256 + threadIdx.x;
  if (i >= n) return;
  const float c = coef[i] * num[i] / (den[i] + NMF_EPS);
  coef[i] = c;
  cb[i] = (bf16_t)c;
  const int r = i & 63, nn = (i >> 6) & 2047, b = i >> 17;
  cT[((size_t)b * 64 + r) * 2048 + nn] = (bf16_t)c;
}

__global__ void k_update_bases(float* __restrict__ bw, const float* __restrict__ num,
                               const float* __restrict__ den,
                               bf16_t* __restrict__ bb, bf16_t* __restrict__ bT, int n) {
  const int i = blockIdx.x * 256 + threadIdx.x;
  if (i >= n) return;
  const float c = bw[i] * num[i] / (den[i] + NMF_EPS);
  bw[i] = c;
  bb[i] = (bf16_t)c;
  const int r = i & 63, d = (i >> 6) & 1023, b = i >> 16;
  bT[((size_t)b * 64 + r) * 1024 + d] = (bf16_t)c;
}

__global__ void k_init_bases(const float* __restrict__ bin, float* __restrict__ bw,
                             bf16_t* __restrict__ bb, bf16_t* __restrict__ bT, int n) {
  const int i = blockIdx.x * 256 + threadIdx.x;
  if (i >= n) return;
  const float v = bin[i];
  bw[i] = v;
  bb[i] = (bf16_t)v;
  const int r = i & 63, d = (i >> 6) & 1023, b = i >> 16;
  bT[((size_t)b * 64 + r) * 1024 + d] = (bf16_t)v;
}

__global__ void k_cvt(const float* __restrict__ in, bf16_t* __restrict__ outp, int n) {
  const int i = blockIdx.x * 256 + threadIdx.x;
  if (i < n) outp[i] = (bf16_t)in[i];
}

// ---------------------------------------------------------------------------
extern "C" void kernel_launch(void* const* d_in, const int* in_sizes, int n_in,
                              void* d_out, int out_size, void* d_ws, size_t ws_size,
                              hipStream_t stream) {
  const float* x        = (const float*)d_in[0];
  const float* Uw       = (const float*)d_in[1];
  const float* Ub       = (const float*)d_in[2];
  const float* Vw       = (const float*)d_in[3];
  const float* Vb       = (const float*)d_in[4];
  const float* g1       = (const float*)d_in[5];
  const float* b1       = (const float*)d_in[6];
  const float* g2       = (const float*)d_in[7];
  const float* b2       = (const float*)d_in[8];
  const float* bases_in = (const float*)d_in[9];
  float* out = (float*)d_out;

  char* ws = (char*)d_ws;
  const size_t MB = 1024ull * 1024ull;
  bf16_t* A1     = (bf16_t*)(ws + 0);         // 32 MB  LN(x) bf16 [16384,1024]
  bf16_t* Uwb    = (bf16_t*)(ws + 32  * MB);  //  8 MB  Uw bf16   [4096,1024]
  bf16_t* Vwb    = (bf16_t*)(ws + 40  * MB);  //  4 MB  Vw bf16   [1024,2048]
  bf16_t* z1     = (bf16_t*)(ws + 44  * MB);  // 64 MB  z1 bf16   [16384,2048]
  float*  z2     = (float*) (ws + 108 * MB);  // 128 MB z2 fp32 (dead after LN2; reused)
  float*  numb   = (float*) (ws + 108 * MB);  //  8 MB  [16,2048,64]
  float*  denb   = (float*) (ws + 116 * MB);  //  8 MB  [16,2048,64]
  float*  num2b  = (float*) (ws + 124 * MB);  //  4 MB  [16,1024,64]
  float*  den2b  = (float*) (ws + 128 * MB);  //  4 MB  [16,1024,64]
  bf16_t* gramb  = (bf16_t*)(ws + 132 * MB);  //  128KB [16,64,64]
  bf16_t* cgramb = (bf16_t*)(ws + 133 * MB);  //  128KB [16,64,64]
  bf16_t* gbuf   = (bf16_t*)(ws + 172 * MB);  // 64 MB  gated act bf16 [16384,2048]
  bf16_t* xn     = (bf16_t*)(ws + 236 * MB);  // 64 MB  [16,1024,2048] (D-major)
  bf16_t* xnT    = (bf16_t*)(ws + 300 * MB);  // 64 MB  [16,2048,1024] (N-major)
  float*  basesw = (float*) (ws + 364 * MB);  //  4 MB  [16,1024,64]
  bf16_t* basesb = (bf16_t*)(ws + 368 * MB);  //  2 MB
  bf16_t* basesT = (bf16_t*)(ws + 370 * MB);  //  2 MB  [16,64,1024]
  float*  coefw  = (float*) (ws + 372 * MB);  //  8 MB  [16,2048,64]
  bf16_t* coefb  = (bf16_t*)(ws + 380 * MB);  //  4 MB
  bf16_t* coefT  = (bf16_t*)(ws + 384 * MB);  //  4 MB  [16,64,2048]

  const dim3 blk(256);

  // Stage 1: LN1, weight converts, GEMM1+GELU+split, LN2+ReLU(+transpose)
  k_ln1<<<dim3(16384), blk, 0, stream>>>(x, g1, b1, A1);
  k_cvt<<<dim3((4096 * 1024) / 256), blk, 0, stream>>>(Uw, Uwb, 4096 * 1024);
  k_cvt<<<dim3((1024 * 2048) / 256), blk, 0, stream>>>(Vw, Vwb, 1024 * 2048);
  k_gemm1<<<dim3(64, 128), blk, 0, stream>>>(A1, Uwb, Ub, z1, z2);
  k_ln2<<<dim3(16384), blk, 0, stream>>>(z2, g2, b2, xn, xnT);
  k_init_bases<<<dim3((16 * 1024 * 64) / 256), blk, 0, stream>>>(
      bases_in, basesw, basesb, basesT, 16 * 1024 * 64);

  // Stage 2: coef0 = softmax(xn^T @ bases)
  k_gemm_f32<<<dim3(1, 16, 16), blk, 0, stream>>>(
      xnT, 1024, 2048L * 1024, basesT, 1024, 64L * 1024,
      numb, 64, 2048L * 64, 2048, 1024);
  k_softmax<<<dim3(32768 / 8), blk, 0, stream>>>(numb, coefw, coefb, coefT);

  // Stage 3: 6 no-grad NMF steps + 1 tracked coef update (it==6)
  for (int it = 0; it < 7; ++it) {
    // num = xn^T @ bases    [2048,64] K=1024
    k_gemm_f32<<<dim3(1, 16, 16), blk, 0, stream>>>(
        xnT, 1024, 2048L * 1024, basesT, 1024, 64L * 1024,
        numb, 64, 2048L * 64, 2048, 1024);
    // gram = bases^T @ bases [64,64] K=1024 (bf16 out, reused as B operand)
    k_gemm_bf16<<<dim3(1, 1, 16), blk, 0, stream>>>(
        basesT, 1024, 64L * 1024, basesT, 1024, 64L * 1024,
        gramb, 64, 64L * 64, 64, 1024);
    // den = coef @ gram      [2048,64] K=64 (gram symmetric)
    k_gemm_f32<<<dim3(1, 16, 16), blk, 0, stream>>>(
        coefb, 64, 2048L * 64, gramb, 64, 64L * 64,
        denb, 64, 2048L * 64, 2048, 64);
    k_update_coef<<<dim3((16 * 2048 * 64) / 256), blk, 0, stream>>>(
        coefw, numb, denb, coefb, coefT, 16 * 2048 * 64);
    if (it == 6) break;
    // num2 = xn @ coef       [1024,64] K=2048
    k_gemm_f32<<<dim3(1, 8, 16), blk, 0, stream>>>(
        xn, 2048, 1024L * 2048, coefT, 2048, 64L * 2048,
        num2b, 64, 1024L * 64, 1024, 2048);
    // cgram = coef^T @ coef  [64,64] K=2048
    k_gemm_bf16<<<dim3(1, 1, 16), blk, 0, stream>>>(
        coefT, 2048, 64L * 2048, coefT, 2048, 64L * 2048,
        cgramb, 64, 64L * 64, 64, 2048);
    // den2 = bases @ cgram   [1024,64] K=64
    k_gemm_f32<<<dim3(1, 8, 16), blk, 0, stream>>>(
        basesb, 64, 1024L * 64, cgramb, 64, 64L * 64,
        den2b, 64, 1024L * 64, 1024, 64);
    k_update_bases<<<dim3((16 * 1024 * 64) / 256), blk, 0, stream>>>(
        basesw, num2b, den2b, basesb, basesT, 16 * 1024 * 64);
  }

  // Stage 4: g = z1 * (bases @ coef^T), out = g @ Vw^T + Vb
  k_recon<<<dim3(32, 8, 16), blk, 0, stream>>>(basesb, coefb, z1, gbuf);
  k_final<<<dim3(16, 128), blk, 0, stream>>>(gbuf, Vwb, Vb, out);
}
